// JonbertaSelfAttention_6305011990918
// MI455X (gfx1250) — matrix operand order repaired and checked by reading, not compiled, not run on hardware
//
#include <hip/hip_runtime.h>

// ---------------------------------------------------------------------------
// Problem constants (from reference)
// ---------------------------------------------------------------------------
constexpr int CB  = 4;     // batch
constexpr int CS  = 1024;  // seq
constexpr int CDM = 1024;  // d_model
constexpr int CH  = 16;    // heads
constexpr int CD  = 64;    // head dim
constexpr int CSE = 512;   // encoder seq
constexpr int CDF = 512;   // encoder feature dim
constexpr int CMAXP = 1024;
constexpr float INV_SQRT_D = 0.125f;  // 1/sqrt(64)

typedef __bf16 bf16;
typedef bf16 v16bf __attribute__((ext_vector_type(16)));
typedef bf16 v8bf  __attribute__((ext_vector_type(8)));
typedef float v8f  __attribute__((ext_vector_type(8)));

// ---------------------------------------------------------------------------
// WMMA helpers (CDNA5 v_wmma_f32_16x16x32_bf16, wave32)
//   A (16x32): lane L: row = L&15, K chunk  {hi*16 .. hi*16+15} contiguous
//   B (32x16): lane L: col = L&15, K chunk  {hi*16 .. hi*16+15} contiguous
//   C (16x16): vgpr r, lane L: M = r + (L>=16 ? 8 : 0), N = L&15
// ---------------------------------------------------------------------------
__device__ inline v8f wmma_bf16(v16bf a, v16bf b, v8f c) {
  return __builtin_amdgcn_wmma_f32_16x16x32_bf16(
      false, a, false, b, (short)0, c, false, false);
}

__device__ inline v8f vzero8() {
  v8f z;
#pragma unroll
  for (int i = 0; i < 8; ++i) z[i] = 0.0f;
  return z;
}

__device__ inline v16bf load_A16(const bf16* base, int ld, int lane) {
  int r  = lane & 15;
  int hi = (lane >> 4) & 1;
  const bf16* p = base + (size_t)r * ld + hi * 16;
  v8bf lo = *(const v8bf*)(p);
  v8bf hh = *(const v8bf*)(p + 8);
  v16bf out;
#pragma unroll
  for (int i = 0; i < 8; ++i) { out[i] = lo[i]; out[i + 8] = hh[i]; }
  return out;
}

__device__ inline v16bf load_B16(const bf16* base, int ld, int lane) {
  int n  = lane & 15;
  int hi = (lane >> 4) & 1;
  return *(const v16bf*)(base + (size_t)n * ld + hi * 16);
}

// Wave-private LDS fence: DS ops are per-wave in-order, but cross-lane
// visibility within the wave needs DScnt==0 before dependent reads.
__device__ inline void lds_fence_wave() {
  asm volatile("s_wait_dscnt 0" ::: "memory");
}

// ---------------------------------------------------------------------------
// f32 -> bf16 conversion
// ---------------------------------------------------------------------------
__global__ void cvt_f32_bf16_kernel(const float* __restrict__ src,
                                    bf16* __restrict__ dst, int n) {
  int i = blockIdx.x * blockDim.x + threadIdx.x;
  if (i < n) dst[i] = (bf16)src[i];
}

// ---------------------------------------------------------------------------
// Generic tiled bf16 WMMA GEMM:  C[m,n] = sum_k X[m,k] * W[n,k] + bias[n]
// (torch Linear convention, W is [N,K]).
// MODE 0: write bf16 head-split  [b, h, s, d]     (slen = S or SE)
// MODE 1: write bf16 head-split transposed [b, h, d, s]
// MODE 2: write f32 row-major [M,N] with residual add
// Block: 256 threads = 8 waves in a 4(M) x 2(N) arrangement; each wave
// computes a 16x64 tile; workgroup tile is 64x128.
// ---------------------------------------------------------------------------
template <int MODE>
__global__ __launch_bounds__(256) void gemm_wmma_kernel(
    const bf16* __restrict__ X, const bf16* __restrict__ W,
    const float* __restrict__ bias, const float* __restrict__ resid,
    void* __restrict__ out, int M, int N, int K, int slen) {
  (void)M;
  int lane = threadIdx.x & 31;
  int w    = threadIdx.x >> 5;
  int wm   = w & 3;
  int wn   = w >> 2;
  int row0 = blockIdx.x * 64 + wm * 16;
  int col0 = blockIdx.y * 128 + wn * 64;

  v8f acc[4];
#pragma unroll
  for (int t = 0; t < 4; ++t) acc[t] = vzero8();

  for (int kk = 0; kk < K; kk += 32) {
    const bf16* xa = X + (size_t)row0 * K + kk;
    if (kk + 32 < K) __builtin_prefetch(xa + 32, 0, 1);  // global_prefetch_b8
    v16bf a = load_A16(xa, K, lane);
#pragma unroll
    for (int t = 0; t < 4; ++t) {
      v16bf bt = load_B16(W + (size_t)(col0 + 16 * t) * K + kk, K, lane);
      acc[t] = wmma_bf16(a, bt, acc[t]);
    }
  }

  int n15 = lane & 15;
  int mh  = (lane >> 4) * 8;
#pragma unroll
  for (int t = 0; t < 4; ++t) {
    int n    = col0 + 16 * t + n15;
    float bv = bias[n];
#pragma unroll
    for (int rr = 0; rr < 8; ++rr) {
      int m     = row0 + rr + mh;
      float val = acc[t][rr] + bv;
      if (MODE == 0) {
        int bidx = m / slen, s = m % slen, hh = n >> 6, d = n & 63;
        ((bf16*)out)[(((size_t)(bidx * CH + hh)) * slen + s) * CD + d] =
            (bf16)val;
      } else if (MODE == 1) {
        int bidx = m / slen, s = m % slen, hh = n >> 6, d = n & 63;
        ((bf16*)out)[(((size_t)(bidx * CH + hh)) * CD + d) * slen + s] =
            (bf16)val;
      } else {
        ((float*)out)[(size_t)m * N + n] =
            val + resid[(size_t)m * N + n];
      }
    }
  }
}

// ---------------------------------------------------------------------------
// Fused attention: self-attn with relative_key_query bias (flash-style online
// softmax) + encoder cross-attn branch, writing merged-head ctx [B,S,DM] bf16.
// Grid: B*H*(S/128) blocks, 256 threads (8 waves); wave = 16 queries.
// Position bias per 16x32 key chunk:
//   bias1[l,r] = (q @ E^T)[l, j],  bias2[l,r] = (k @ E^T)[r, j],
//   j = l - r + 1023 lies in a 47-wide band -> 3 WMMA B-tiles of E rows,
//   band GEMMs staged to wave-private LDS, diagonal-gathered per element.
// ---------------------------------------------------------------------------
__global__ __launch_bounds__(256) void attn_relposbias_kernel(
    const bf16* __restrict__ q, const bf16* __restrict__ k,
    const bf16* __restrict__ vT, const bf16* __restrict__ fk,
    const bf16* __restrict__ fvT, const bf16* __restrict__ E,
    const float* __restrict__ mask, bf16* __restrict__ ctx) {
  __shared__ float ldsC1[8][16 * 48];             // q @ E_band^T
  __shared__ float ldsC2[8][32 * 48];             // k @ E_band^T
  __shared__ __align__(32) bf16 ldsP[8][16 * 32]; // probs staging

  int lane = threadIdx.x & 31;
  int wave = threadIdx.x >> 5;
  int n15  = lane & 15;
  int hi   = (lane >> 4) & 1;
  int mh   = hi * 8;

  int lblk = blockIdx.x & 7;
  int bh   = blockIdx.x >> 3;   // b*H + h
  int b    = bh / CH;
  int h    = bh % CH;
  int l0   = lblk * 128 + wave * 16;

  // Q tile 16x64 as two A operands (K slices 0..31, 32..63)
  const bf16* qbase = q + ((size_t)bh * CS + l0) * CD;
  v16bf aq0 = load_A16(qbase, CD, lane);
  v16bf aq1 = load_A16(qbase + 32, CD, lane);

  v8f acc[4];
#pragma unroll
  for (int t = 0; t < 4; ++t) acc[t] = vzero8();
  float mrow[8], lrow[8];
#pragma unroll
  for (int rr = 0; rr < 8; ++rr) { mrow[rr] = -1e30f; lrow[rr] = 0.0f; }

  float* C1 = &ldsC1[wave][0];
  float* C2 = &ldsC2[wave][0];
  bf16*  P  = &ldsP[wave][0];

  // ---------------- self-attention over S keys, 32 per chunk --------------
  for (int r0 = 0; r0 < CS; r0 += 32) {
    const bf16* kb = k + ((size_t)bh * CS + r0) * CD;

    // scores: q(16x64) @ k^T(64x32) -> two 16x16 tiles
    v8f s0 = vzero8(), s1 = vzero8();
    s0 = wmma_bf16(aq0, load_B16(kb, CD, lane), s0);
    s0 = wmma_bf16(aq1, load_B16(kb + 32, CD, lane), s0);
    s1 = wmma_bf16(aq0, load_B16(kb + 16 * CD, CD, lane), s1);
    s1 = wmma_bf16(aq1, load_B16(kb + 16 * CD + 32, CD, lane), s1);

    // dist_emb band: j = l - r + 1023, c = (l-l0) - (r-r0) + 31 in [0,46]
    int jb = l0 - r0 - 31 + (CMAXP - 1);
    v16bf bE[3][2];
#pragma unroll
    for (int t = 0; t < 3; ++t) {
      int er = jb + 16 * t + n15;
      er = er < 0 ? 0 : er;
      er = er > (2 * CMAXP - 2) ? (2 * CMAXP - 2) : er;
      const bf16* ep = E + (size_t)er * CD + hi * 16;
      bE[t][0] = *(const v16bf*)ep;
      bE[t][1] = *(const v16bf*)(ep + 32);
    }
    // C1 = q_tile @ E_band^T  (16x48)
#pragma unroll
    for (int t = 0; t < 3; ++t) {
      v8f c1 = vzero8();
      c1 = wmma_bf16(aq0, bE[t][0], c1);
      c1 = wmma_bf16(aq1, bE[t][1], c1);
#pragma unroll
      for (int rr = 0; rr < 8; ++rr)
        C1[(rr + mh) * 48 + 16 * t + n15] = c1[rr];
    }
    // C2 = k_chunk @ E_band^T  (32x48)
#pragma unroll
    for (int mt = 0; mt < 2; ++mt) {
      v16bf ak0 = load_A16(kb + (size_t)16 * mt * CD, CD, lane);
      v16bf ak1 = load_A16(kb + (size_t)16 * mt * CD + 32, CD, lane);
#pragma unroll
      for (int t = 0; t < 3; ++t) {
        v8f c2 = vzero8();
        c2 = wmma_bf16(ak0, bE[t][0], c2);
        c2 = wmma_bf16(ak1, bE[t][1], c2);
#pragma unroll
        for (int rr = 0; rr < 8; ++rr)
          C2[(16 * mt + rr + mh) * 48 + 16 * t + n15] = c2[rr];
      }
    }
    lds_fence_wave();

    float m0v = mask[(size_t)b * CS + r0 + n15];
    float m1v = mask[(size_t)b * CS + r0 + 16 + n15];
    float sc0[8], sc1[8];
#pragma unroll
    for (int rr = 0; rr < 8; ++rr) {
      int m  = rr + mh;
      int c0 = m - n15 + 31;  // key tile 0 (r_loc = n)
      float bias0 = C1[m * 48 + c0] + C2[n15 * 48 + c0];
      sc0[rr] = (s0[rr] + bias0) * INV_SQRT_D + m0v;
      int c1i = m - n15 + 15; // key tile 1 (r_loc = 16+n)
      float bias1 = C1[m * 48 + c1i] + C2[(16 + n15) * 48 + c1i];
      sc1[rr] = (s1[rr] + bias1) * INV_SQRT_D + m1v;
    }

    // online softmax update (row reduce = 16-lane xor-shuffle group)
#pragma unroll
    for (int rr = 0; rr < 8; ++rr) {
      float mx = fmaxf(sc0[rr], sc1[rr]);
#pragma unroll
      for (int off = 1; off < 16; off <<= 1)
        mx = fmaxf(mx, __shfl_xor(mx, off, 32));
      float mnew  = fmaxf(mrow[rr], mx);
      float alpha = __expf(mrow[rr] - mnew);
      float p0 = __expf(sc0[rr] - mnew);
      float p1 = __expf(sc1[rr] - mnew);
      float rs = p0 + p1;
#pragma unroll
      for (int off = 1; off < 16; off <<= 1) rs += __shfl_xor(rs, off, 32);
      lrow[rr] = lrow[rr] * alpha + rs;
      mrow[rr] = mnew;
#pragma unroll
      for (int t = 0; t < 4; ++t) acc[t][rr] *= alpha;
      int m = rr + mh;
      P[m * 32 + n15]      = (bf16)p0;
      P[m * 32 + 16 + n15] = (bf16)p1;
    }
    lds_fence_wave();

    // PV: probs(16x32) @ V(32x64), V stored transposed [b,h,d,s]
    v16bf ap = load_A16(P, 32, lane);
    const bf16* vb = vT + (size_t)bh * CD * CS + r0;
#pragma unroll
    for (int t = 0; t < 4; ++t) {
      v16bf bv = load_B16(vb + (size_t)16 * t * CS, CS, lane);
      acc[t] = wmma_bf16(ap, bv, acc[t]);
    }
  }

  // finalize self-attn context
  float ctxs[4][8];
#pragma unroll
  for (int t = 0; t < 4; ++t)
#pragma unroll
    for (int rr = 0; rr < 8; ++rr) ctxs[t][rr] = acc[t][rr] / lrow[rr];

  // ---------------- encoder cross-attention branch -------------------------
#pragma unroll
  for (int t = 0; t < 4; ++t) acc[t] = vzero8();
#pragma unroll
  for (int rr = 0; rr < 8; ++rr) { mrow[rr] = -1e30f; lrow[rr] = 0.0f; }

  for (int r0 = 0; r0 < CSE; r0 += 32) {
    const bf16* kb = fk + ((size_t)bh * CSE + r0) * CD;
    v8f s0 = vzero8(), s1 = vzero8();
    s0 = wmma_bf16(aq0, load_B16(kb, CD, lane), s0);
    s0 = wmma_bf16(aq1, load_B16(kb + 32, CD, lane), s0);
    s1 = wmma_bf16(aq0, load_B16(kb + 16 * CD, CD, lane), s1);
    s1 = wmma_bf16(aq1, load_B16(kb + 16 * CD + 32, CD, lane), s1);

#pragma unroll
    for (int rr = 0; rr < 8; ++rr) {
      float sc0 = s0[rr] * INV_SQRT_D;
      float sc1 = s1[rr] * INV_SQRT_D;
      float mx = fmaxf(sc0, sc1);
#pragma unroll
      for (int off = 1; off < 16; off <<= 1)
        mx = fmaxf(mx, __shfl_xor(mx, off, 32));
      float mnew  = fmaxf(mrow[rr], mx);
      float alpha = __expf(mrow[rr] - mnew);
      float p0 = __expf(sc0 - mnew);
      float p1 = __expf(sc1 - mnew);
      float rs = p0 + p1;
#pragma unroll
      for (int off = 1; off < 16; off <<= 1) rs += __shfl_xor(rs, off, 32);
      lrow[rr] = lrow[rr] * alpha + rs;
      mrow[rr] = mnew;
#pragma unroll
      for (int t = 0; t < 4; ++t) acc[t][rr] *= alpha;
      int m = rr + mh;
      P[m * 32 + n15]      = (bf16)p0;
      P[m * 32 + 16 + n15] = (bf16)p1;
    }
    lds_fence_wave();

    v16bf ap = load_A16(P, 32, lane);
    const bf16* vb = fvT + (size_t)bh * CD * CSE + r0;
#pragma unroll
    for (int t = 0; t < 4; ++t) {
      v16bf bv = load_B16(vb + (size_t)16 * t * CSE, CSE, lane);
      acc[t] = wmma_bf16(ap, bv, acc[t]);
    }
    lds_fence_wave();  // protect P before next chunk overwrites
  }

#pragma unroll
  for (int t = 0; t < 4; ++t)
#pragma unroll
    for (int rr = 0; rr < 8; ++rr) ctxs[t][rr] += acc[t][rr] / lrow[rr];

  // merged-head store: ctx[b, l0+m, h*64 + 16t + n] (bf16)
#pragma unroll
  for (int t = 0; t < 4; ++t) {
    int col = h * CD + 16 * t + n15;
#pragma unroll
    for (int rr = 0; rr < 8; ++rr) {
      int m = rr + mh;
      ctx[((size_t)b * CS + l0 + m) * CDM + col] = (bf16)ctxs[t][rr];
    }
  }
}

// ---------------------------------------------------------------------------
// LayerNorm over last dim (1024): one block per row.
// ---------------------------------------------------------------------------
__global__ __launch_bounds__(256) void layernorm_kernel(
    const float* __restrict__ y, const float* __restrict__ g,
    const float* __restrict__ be, float* __restrict__ out) {
  __shared__ float red[8];
  int row  = blockIdx.x;
  int tid  = threadIdx.x;
  int lane = tid & 31;
  int w    = tid >> 5;
  const float* yr = y + (size_t)row * CDM;

  float v[4];
  float s = 0.0f;
#pragma unroll
  for (int i = 0; i < 4; ++i) { v[i] = yr[tid + 256 * i]; s += v[i]; }
#pragma unroll
  for (int off = 1; off < 32; off <<= 1) s += __shfl_xor(s, off, 32);
  if (lane == 0) red[w] = s;
  __syncthreads();
  float tot = 0.0f;
#pragma unroll
  for (int i = 0; i < 8; ++i) tot += red[i];
  float mu = tot * (1.0f / CDM);
  __syncthreads();

  float s2 = 0.0f;
#pragma unroll
  for (int i = 0; i < 4; ++i) { float d = v[i] - mu; s2 += d * d; }
#pragma unroll
  for (int off = 1; off < 32; off <<= 1) s2 += __shfl_xor(s2, off, 32);
  if (lane == 0) red[w] = s2;
  __syncthreads();
  float tv = 0.0f;
#pragma unroll
  for (int i = 0; i < 8; ++i) tv += red[i];
  float rstd = rsqrtf(tv * (1.0f / CDM) + 1e-12f);

#pragma unroll
  for (int i = 0; i < 4; ++i) {
    int c = tid + 256 * i;
    out[(size_t)row * CDM + c] = (v[i] - mu) * rstd * g[c] + be[c];
  }
}

// ---------------------------------------------------------------------------
// Workspace layout (bytes), ~76 MB total
// ---------------------------------------------------------------------------
constexpr size_t OFF_HSB  = 0;                        // hidden bf16 [4096,1024]
constexpr size_t OFF_ENCB = OFF_HSB + 8388608;        // encoder bf16 [2048,512]
constexpr size_t OFF_WQB  = OFF_ENCB + 2097152;
constexpr size_t OFF_WKB  = OFF_WQB + 2097152;
constexpr size_t OFF_WVB  = OFF_WKB + 2097152;
constexpr size_t OFF_WFKB = OFF_WVB + 2097152;
constexpr size_t OFF_WFVB = OFF_WFKB + 1048576;
constexpr size_t OFF_WOB  = OFF_WFVB + 1048576;
constexpr size_t OFF_EB   = OFF_WOB + 2097152;        // dist_emb bf16 (2048 rows)
constexpr size_t OFF_QB   = OFF_EB + 262144;          // q  [B,H,S,D] bf16
constexpr size_t OFF_KB   = OFF_QB + 8388608;         // k  [B,H,S,D] bf16
constexpr size_t OFF_VTB  = OFF_KB + 8388608;         // vT [B,H,D,S] bf16
constexpr size_t OFF_FKB  = OFF_VTB + 8388608;        // fk [B,H,SE,D] bf16
constexpr size_t OFF_FVTB = OFF_FKB + 4194304;        // fvT [B,H,D,SE] bf16
constexpr size_t OFF_CTXB = OFF_FVTB + 4194304;       // ctx [B,S,DM] bf16
constexpr size_t OFF_YF   = OFF_CTXB + 8388608;       // y f32 [B,S,DM]

extern "C" void kernel_launch(void* const* d_in, const int* in_sizes, int n_in,
                              void* d_out, int out_size, void* d_ws,
                              size_t ws_size, hipStream_t stream) {
  (void)in_sizes; (void)n_in; (void)out_size; (void)ws_size;
  const float* hs   = (const float*)d_in[0];
  const float* mask = (const float*)d_in[1];
  const float* enc  = (const float*)d_in[2];
  const float* Wq   = (const float*)d_in[3];
  const float* bq   = (const float*)d_in[4];
  const float* Wk   = (const float*)d_in[5];
  const float* bk   = (const float*)d_in[6];
  const float* Wv   = (const float*)d_in[7];
  const float* bv   = (const float*)d_in[8];
  const float* Wfk  = (const float*)d_in[9];
  const float* bfk  = (const float*)d_in[10];
  const float* Wfv  = (const float*)d_in[11];
  const float* bfv  = (const float*)d_in[12];
  const float* dist = (const float*)d_in[13];
  const float* Wo   = (const float*)d_in[14];
  const float* bo   = (const float*)d_in[15];
  const float* ln_g = (const float*)d_in[16];
  const float* ln_b = (const float*)d_in[17];

  char* ws = (char*)d_ws;
  bf16* hsb  = (bf16*)(ws + OFF_HSB);
  bf16* encb = (bf16*)(ws + OFF_ENCB);
  bf16* wqb  = (bf16*)(ws + OFF_WQB);
  bf16* wkb  = (bf16*)(ws + OFF_WKB);
  bf16* wvb  = (bf16*)(ws + OFF_WVB);
  bf16* wfkb = (bf16*)(ws + OFF_WFKB);
  bf16* wfvb = (bf16*)(ws + OFF_WFVB);
  bf16* wob  = (bf16*)(ws + OFF_WOB);
  bf16* eb   = (bf16*)(ws + OFF_EB);
  bf16* qb   = (bf16*)(ws + OFF_QB);
  bf16* kb   = (bf16*)(ws + OFF_KB);
  bf16* vtb  = (bf16*)(ws + OFF_VTB);
  bf16* fkb  = (bf16*)(ws + OFF_FKB);
  bf16* fvtb = (bf16*)(ws + OFF_FVTB);
  bf16* ctxb = (bf16*)(ws + OFF_CTXB);
  float* yf  = (float*)(ws + OFF_YF);

  auto cvt = [&](const float* s, bf16* d, int n) {
    cvt_f32_bf16_kernel<<<(n + 255) / 256, 256, 0, stream>>>(s, d, n);
  };
  cvt(hs, hsb, CB * CS * CDM);
  cvt(enc, encb, CB * CSE * CDF);
  cvt(Wq, wqb, CDM * CDM);
  cvt(Wk, wkb, CDM * CDM);
  cvt(Wv, wvb, CDM * CDM);
  cvt(Wfk, wfkb, CDM * CDF);
  cvt(Wfv, wfvb, CDM * CDF);
  cvt(Wo, wob, CDM * CDM);
  cvt(dist, eb, (2 * CMAXP - 1) * CD);

  // Projections: Q/K head-split, V transposed; feature K/V likewise.
  dim3 gQKV(CB * CS / 64, CDM / 128);
  gemm_wmma_kernel<0><<<gQKV, 256, 0, stream>>>(hsb, wqb, bq, nullptr, qb,
                                                CB * CS, CDM, CDM, CS);
  gemm_wmma_kernel<0><<<gQKV, 256, 0, stream>>>(hsb, wkb, bk, nullptr, kb,
                                                CB * CS, CDM, CDM, CS);
  gemm_wmma_kernel<1><<<gQKV, 256, 0, stream>>>(hsb, wvb, bv, nullptr, vtb,
                                                CB * CS, CDM, CDM, CS);
  dim3 gF(CB * CSE / 64, CDM / 128);
  gemm_wmma_kernel<0><<<gF, 256, 0, stream>>>(encb, wfkb, bfk, nullptr, fkb,
                                              CB * CSE, CDM, CDF, CSE);
  gemm_wmma_kernel<1><<<gF, 256, 0, stream>>>(encb, wfvb, bfv, nullptr, fvtb,
                                              CB * CSE, CDM, CDF, CSE);

  // Fused dual-branch attention with relative-position bias.
  attn_relposbias_kernel<<<CB * CH * (CS / 128), 256, 0, stream>>>(
      qb, kb, vtb, fkb, fvtb, eb, mask, ctxb);

  // Output projection + residual, then LayerNorm.
  gemm_wmma_kernel<2><<<gQKV, 256, 0, stream>>>(ctxb, wob, bo, hs, yf,
                                                CB * CS, CDM, CDM, CS);
  layernorm_kernel<<<CB * CS, 256, 0, stream>>>(yf, ln_g, ln_b,
                                                (float*)d_out);
}
